// OTGraphMatcher_15771119911319
// MI455X (gfx1250) — compile-verified, matrix operand order checked by reading, not run on hardware
//
#include <hip/hip_runtime.h>

typedef unsigned short u16;
typedef __attribute__((ext_vector_type(16))) __bf16 v16bf;
typedef __attribute__((ext_vector_type(8)))  float  v8f;
typedef int v4i __attribute__((vector_size(16)));
typedef __attribute__((address_space(1))) void gas_void;
typedef __attribute__((address_space(3))) void las_void;
typedef __attribute__((address_space(1))) v4i  gas_v4i;
typedef __attribute__((address_space(3))) v4i  las_v4i;

constexpr int   N4      = 4096;
constexpr int   HID     = 1024;
constexpr int   EMB     = 256;
constexpr float REG_INV = 10.0f;   // 1/reg
constexpr int   SINK_IT = 50;

// GEMM epilogue modes
constexpr int EPI_H         = 0;  // outH[m][n] = bf16(acc)
constexpr int EPI_BIAS_RELU = 1;  // outH[m][n] = bf16(relu(acc + bias[n]))
constexpr int EPI_BIAS_BOTH = 2;  // outF[m][n] = acc + bias[n]; outH = bf16(same)
constexpr int EPI_KEXP      = 3;  // c = max(n1[m]+n2[n]-2acc,0); k=exp(-c/reg) -> outF[m][n], Kt[n][m]
constexpr int EPI_LOSS      = 4;  // atomicAdd(lossAcc, sum (Mref[m][n]-acc)^2)
constexpr int EPI_HT        = 5;  // outH[n][m] = bf16(acc)   (transposed bf16 output)

__device__ __forceinline__ u16 f2bf(float f) {
    unsigned int u = __float_as_uint(f);
    unsigned int r = (u + 0x7fffu + ((u >> 16) & 1u)) >> 16;   // round-to-nearest-even
    return (u16)r;
}

// CDNA5 async global->LDS copy (ASYNCcnt-tracked, no VGPR round trip).
__device__ __forceinline__ void async_b128(const u16* g, u16* l) {
#if __has_builtin(__builtin_amdgcn_global_load_async_to_lds_b128)
    __builtin_amdgcn_global_load_async_to_lds_b128(
        (gas_v4i*)(gas_void*)const_cast<u16*>(g),
        (las_v4i*)(las_void*)l, 0, 0);
#else
    *(uint4*)l = *(const uint4*)g;   // sync fallback
#endif
}

template <int N>
__device__ __forceinline__ void wait_async() {
#if __has_builtin(__builtin_amdgcn_s_wait_asynccnt)
    __builtin_amdgcn_s_wait_asynccnt((unsigned short)N);
#else
    asm volatile("s_wait_asynccnt %0" :: "n"(N) : "memory");
#endif
}

// ---------------------------------------------------------------------------
// Small helper kernels
// ---------------------------------------------------------------------------

// dinv[row] = rsqrt(max(#nonzero(M[row,:]) + 1 (self loop), 1e-12))
__global__ void otgm_rowsum(const float* __restrict__ M, float* __restrict__ dinv) {
    __shared__ float red[256];
    const int row = blockIdx.x;
    const float* r = M + (size_t)row * N4;
    float s = 0.f;
    for (int j = threadIdx.x; j < N4; j += 256) s += (r[j] != 0.f) ? 1.f : 0.f;
    red[threadIdx.x] = s; __syncthreads();
    for (int off = 128; off > 0; off >>= 1) {
        if (threadIdx.x < off) red[threadIdx.x] += red[threadIdx.x + off];
        __syncthreads();
    }
    if (threadIdx.x == 0) dinv[row] = rsqrtf(fmaxf(red[0] + 1.f, 1e-12f));
}

// Ahat[i][j] = ((M!=0)+I) * dinv[i]*dinv[j]  -> bf16
__global__ void otgm_ahat(const float* __restrict__ M, const float* __restrict__ dinv,
                          u16* __restrict__ out) {
    size_t idx = (size_t)blockIdx.x * 256 + threadIdx.x;
    int i = (int)(idx >> 12), j = (int)(idx & (N4 - 1));
    float a = ((M[idx] != 0.f) ? 1.f : 0.f) + ((i == j) ? 1.f : 0.f);
    out[idx] = f2bf(a * dinv[i] * dinv[j]);
}

// out[c][r] = bf16(in[r][c])  (tiled transpose, in is [rows x cols] f32)
__global__ void otgm_tobf16_T(const float* __restrict__ in, u16* __restrict__ out,
                              int rows, int cols) {
    __shared__ float tile[32][33];
    const int bx = blockIdx.x * 32;          // column base
    const int by = blockIdx.y * 32;          // row base
    const int tx = threadIdx.x & 31, ty = threadIdx.x >> 5;   // 32 x 8
#pragma unroll
    for (int i = ty; i < 32; i += 8)
        tile[i][tx] = in[(size_t)(by + i) * cols + bx + tx];
    __syncthreads();
#pragma unroll
    for (int i = ty; i < 32; i += 8)
        out[(size_t)(bx + i) * rows + by + tx] = f2bf(tile[tx][i]);
}

// nrm[row] = sum_j Z[row][j]^2   (Z is [rows x EMB], EMB == blockDim 256)
__global__ void otgm_rownorm(const float* __restrict__ Z, float* __restrict__ nrm) {
    __shared__ float red[256];
    const int row = blockIdx.x;
    float v = Z[(size_t)row * EMB + threadIdx.x];
    red[threadIdx.x] = v * v; __syncthreads();
    for (int off = 128; off > 0; off >>= 1) {
        if (threadIdx.x < off) red[threadIdx.x] += red[threadIdx.x + off];
        __syncthreads();
    }
    if (threadIdx.x == 0) nrm[row] = red[0];
}

__global__ void otgm_fill(float* __restrict__ p, int n, float val) {
    int i = blockIdx.x * 256 + threadIdx.x;
    if (i < n) p[i] = val;
}

// vout[row] = numer / dot(Km[row,:], vin)     (block per row; K is L2-resident)
__global__ void otgm_matvec(const float* __restrict__ Km, const float* __restrict__ vin,
                            float* __restrict__ vout, float numer) {
    __shared__ float red[256];
    const int row = blockIdx.x;
    const float4* kr = (const float4*)(Km + (size_t)row * N4);
    const float4* vv = (const float4*)vin;
    float s = 0.f;
    for (int j = threadIdx.x; j < N4 / 4; j += 256) {
        float4 k4 = kr[j]; float4 v4 = vv[j];
        s += k4.x * v4.x + k4.y * v4.y + k4.z * v4.z + k4.w * v4.w;
    }
    red[threadIdx.x] = s; __syncthreads();
    for (int off = 128; off > 0; off >>= 1) {
        if (threadIdx.x < off) red[threadIdx.x] += red[threadIdx.x + off];
        __syncthreads();
    }
    if (threadIdx.x == 0) vout[row] = numer / red[0];
}

// S = u[i] * K[i][j] * v[j]  ->  f32 (d_out) and bf16 (for final GEMMs)
__global__ void otgm_build_S(const float* __restrict__ Km, const float* __restrict__ u,
                             const float* __restrict__ v, float* __restrict__ Sout,
                             u16* __restrict__ Sh) {
    size_t idx = (size_t)blockIdx.x * 256 + threadIdx.x;
    int i = (int)(idx >> 12), j = (int)(idx & (N4 - 1));
    float s = u[i] * Km[idx] * v[j];
    Sout[idx] = s;
    Sh[idx] = f2bf(s);
}

__global__ void otgm_finalize(const float* __restrict__ acc, float* __restrict__ out) {
    out[0] = sqrtf(acc[0]);
}

// ---------------------------------------------------------------------------
// bf16 WMMA GEMM:  C[M x N] (f32 acc) = A[M x K] @ Bt^T, Bt is [N x K] row-major.
// 128x128 block tile, BK = 32, 256 threads = 8 waves (2 x 4 wave grid),
// each wave owns 64x32 = 4x2 WMMA 16x16 tiles.
// Double-buffered LDS; tiles staged with async global->LDS copies (ASYNCcnt).
// ---------------------------------------------------------------------------
#define BM  128
#define BN  128
#define BK  32
#define LDT 40   // LDS row stride (bf16 elems); row pitch 80 B = 5 x 16 B, conflict-skewed

template <int EPI>
__global__ __launch_bounds__(256, 2)
void otgm_gemm_bf16(const u16* __restrict__ A, const u16* __restrict__ Bt,
                    int M, int N, int K,
                    float* __restrict__ outF, u16* __restrict__ outH,
                    const float* __restrict__ bias,
                    const float* __restrict__ n1, const float* __restrict__ n2,
                    float* __restrict__ Kt, const float* __restrict__ Mref,
                    float* __restrict__ lossAcc) {
    __shared__ __align__(16) u16 As[2][BM * LDT];
    __shared__ __align__(16) u16 Bs[2][BN * LDT];

    const int tid  = threadIdx.x;
    const int bM   = blockIdx.x * BM;
    const int bN   = blockIdx.y * BN;
    const int lane = tid & 31;
    const int w    = tid >> 5;
    const int wm   = w & 1;        // 0..1 : 64-row stripe
    const int wn   = w >> 1;       // 0..3 : 32-col stripe
    const int half = lane >> 4;    // K-half selector per ISA 16-bit A layout
    const int r    = lane & 15;    // row (A) / col (B) within 16

    // per-thread staging coordinates: 2 chunks of 8 bf16 per matrix per tile
    const int c0row = tid >> 2, c0kc = tid & 3;
    const int c1row = (tid + 256) >> 2, c1kc = tid & 3;   // (tid+256)&3 == tid&3

    // issue async copies for K-tile k0 into buffer `buf` (4 async ops / thread)
    auto stage = [&](int buf, int k0) {
        async_b128(A  + (size_t)(bM + c0row) * K + k0 + c0kc * 8, &As[buf][c0row * LDT + c0kc * 8]);
        async_b128(A  + (size_t)(bM + c1row) * K + k0 + c1kc * 8, &As[buf][c1row * LDT + c1kc * 8]);
        async_b128(Bt + (size_t)(bN + c0row) * K + k0 + c0kc * 8, &Bs[buf][c0row * LDT + c0kc * 8]);
        async_b128(Bt + (size_t)(bN + c1row) * K + k0 + c1kc * 8, &Bs[buf][c1row * LDT + c1kc * 8]);
    };

    v8f acc[4][2];
    const v8f vzero = {0.f, 0.f, 0.f, 0.f, 0.f, 0.f, 0.f, 0.f};
#pragma unroll
    for (int mt = 0; mt < 4; ++mt)
#pragma unroll
        for (int nt = 0; nt < 2; ++nt) acc[mt][nt] = vzero;

    const int T = K / BK;
    stage(0, 0);
    for (int t = 0; t < T; ++t) {
        const int buf = t & 1;
        if (t + 1 < T) {
            stage(buf ^ 1, (t + 1) * BK);     // prefetch next tile into other buffer
            wait_async<4>();                  // wait only for *current* tile's batch
        } else {
            wait_async<0>();
        }
        // long-range L2 prefetch, two tiles ahead (gfx1250 global_prefetch)
        if (t + 2 < T) {
            if (tid < 128) __builtin_prefetch(A  + (size_t)(bM + tid) * K + (t + 2) * BK, 0, 1);
            else           __builtin_prefetch(Bt + (size_t)(bN + tid - 128) * K + (t + 2) * BK, 0, 1);
        }
        __syncthreads();   // current tile visible to all waves

        // fragments: VGPR0..3 = K half*8+0..7, VGPR4..7 = K 16+half*8+0..7
        v16bf afrag[4], bfrag[2];
#pragma unroll
        for (int mt = 0; mt < 4; ++mt) {
            const u16* p = &As[buf][(wm * 64 + mt * 16 + r) * LDT + half * 8];
            ((uint4*)&afrag[mt])[0] = *(const uint4*)p;
            ((uint4*)&afrag[mt])[1] = *(const uint4*)(p + 16);
        }
#pragma unroll
        for (int nt = 0; nt < 2; ++nt) {
            const u16* p = &Bs[buf][(wn * 32 + nt * 16 + r) * LDT + half * 8];
            ((uint4*)&bfrag[nt])[0] = *(const uint4*)p;
            ((uint4*)&bfrag[nt])[1] = *(const uint4*)(p + 16);
        }
#pragma unroll
        for (int mt = 0; mt < 4; ++mt)
#pragma unroll
            for (int nt = 0; nt < 2; ++nt)
                acc[mt][nt] = __builtin_amdgcn_wmma_f32_16x16x32_bf16(
                    false, afrag[mt], false, bfrag[nt], (short)0, acc[mt][nt],
                    false, false);
        __syncthreads();   // all reads of `buf` done before it is overwritten
    }

    // ---- epilogue: C/D layout — lane n = lane&15, m = (lane>>4)*8 + vgpr ----
    float lsum = 0.f;
#pragma unroll
    for (int mt = 0; mt < 4; ++mt) {
#pragma unroll
        for (int nt = 0; nt < 2; ++nt) {
            const int gn     = bN + wn * 32 + nt * 16 + (lane & 15);
            const int gmBase = bM + wm * 64 + mt * 16 + (lane >> 4) * 8;
#pragma unroll
            for (int vi = 0; vi < 8; ++vi) {
                const int gm = gmBase + vi;
                float val = acc[mt][nt][vi];
                if (EPI == EPI_H) {
                    outH[(size_t)gm * N + gn] = f2bf(val);
                } else if (EPI == EPI_HT) {
                    outH[(size_t)gn * M + gm] = f2bf(val);
                } else if (EPI == EPI_BIAS_RELU) {
                    outH[(size_t)gm * N + gn] = f2bf(fmaxf(val + bias[gn], 0.f));
                } else if (EPI == EPI_BIAS_BOTH) {
                    float x = val + bias[gn];
                    outF[(size_t)gm * N + gn] = x;
                    outH[(size_t)gm * N + gn] = f2bf(x);
                } else if (EPI == EPI_KEXP) {
                    float c  = fmaxf(n1[gm] + n2[gn] - 2.f * val, 0.f);
                    float kv = __expf(-c * REG_INV);
                    outF[(size_t)gm * N + gn] = kv;       // K  row-major
                    Kt[(size_t)gn * M + gm]   = kv;       // K^T row-major
                } else { // EPI_LOSS
                    float d = Mref[(size_t)gm * N + gn] - val;
                    lsum += d * d;
                }
            }
        }
    }
    if (EPI == EPI_LOSS) {
#pragma unroll
        for (int off = 16; off > 0; off >>= 1) lsum += __shfl_xor(lsum, off);
        if (lane == 0) atomicAdd(lossAcc, lsum);
    }
}

// ---------------------------------------------------------------------------
// Host orchestration
// ---------------------------------------------------------------------------
extern "C" void kernel_launch(void* const* d_in, const int* in_sizes, int n_in,
                              void* d_out, int out_size, void* d_ws, size_t ws_size,
                              hipStream_t stream) {
    const float* M1 = (const float*)d_in[0];
    const float* M2 = (const float*)d_in[1];
    const float* W1 = (const float*)d_in[2];
    const float* b1 = (const float*)d_in[3];
    const float* W2 = (const float*)d_in[4];
    const float* b2 = (const float*)d_in[5];

    float* outLoss = (float*)d_out;
    float* outS    = outLoss + 1;

    char* wsb = (char*)d_ws;
    const size_t MB = 1ull << 20;
    float* Kmat = (float*)(wsb + 0 * MB);      //  64 MB  f32 K
    float* Ktm  = (float*)(wsb + 64 * MB);     //  64 MB  f32 K^T
    u16* A1h    = (u16*)(wsb + 128 * MB);      //  32 MB  Ahat1 bf16  (reused as S bf16)
    u16* A2h    = (u16*)(wsb + 160 * MB);      //  32 MB  Ahat2 bf16  (reused as T = S@M2 bf16)
    u16* M2t    = (u16*)(wsb + 192 * MB);      //  32 MB  M2^T bf16
    u16* W1t    = (u16*)(wsb + 224 * MB);      //   8 MB  W1^T bf16 [HID x N4]
    u16* W2t    = (u16*)(wsb + 232 * MB);      // 0.5 MB  W2^T bf16 [EMB x HID]
    u16* Hh     = (u16*)(wsb + 233 * MB);      //   8 MB  hidden bf16 [N4 x HID]
    u16* Gt     = (u16*)(wsb + 241 * MB);      //   2 MB  (H@W2)^T bf16 [EMB x N4]
    float* Z1f  = (float*)(wsb + 243 * MB);    //   4 MB
    float* Z2f  = (float*)(wsb + 247 * MB);    //   4 MB
    u16* Z1h    = (u16*)(wsb + 251 * MB);      //   2 MB
    u16* Z2h    = (u16*)(wsb + 253 * MB);      //   2 MB
    float* dinv1   = (float*)(wsb + 255 * MB);
    float* dinv2   = dinv1 + N4;
    float* nn1     = dinv2 + N4;
    float* nn2     = nn1 + N4;
    float* uvec    = nn2 + N4;
    float* vvec    = uvec + N4;
    float* lossAcc = vvec + N4;
    u16* Sh = A1h;   // Ahat1 dead after Z1 GEMM
    u16* Th = A2h;   // Ahat2 dead after Z2 GEMM

    const int ELT_BLK = (N4 * N4) / 256;   // 65536 blocks for N^2 elementwise

    // --- GCN-normalized adjacencies (bf16) + pre-transposed B operands ---
    otgm_rowsum<<<N4, 256, 0, stream>>>(M1, dinv1);
    otgm_rowsum<<<N4, 256, 0, stream>>>(M2, dinv2);
    otgm_ahat<<<ELT_BLK, 256, 0, stream>>>(M1, dinv1, A1h);
    otgm_ahat<<<ELT_BLK, 256, 0, stream>>>(M2, dinv2, A2h);
    otgm_tobf16_T<<<dim3(HID / 32, N4 / 32), 256, 0, stream>>>(W1, W1t, N4, HID);
    otgm_tobf16_T<<<dim3(EMB / 32, HID / 32), 256, 0, stream>>>(W2, W2t, HID, EMB);
    otgm_tobf16_T<<<dim3(N4 / 32, N4 / 32), 256, 0, stream>>>(M2, M2t, N4, N4);

    // --- encoder, graph 1:  H = relu(Ahat1@W1 + b1); Gt = (H@W2)^T; Z1 = Ahat1@G + b2 ---
    otgm_gemm_bf16<EPI_BIAS_RELU><<<dim3(N4 / BM, HID / BN), 256, 0, stream>>>(
        A1h, W1t, N4, HID, N4, nullptr, Hh, b1, nullptr, nullptr, nullptr, nullptr, nullptr);
    otgm_gemm_bf16<EPI_HT><<<dim3(N4 / BM, EMB / BN), 256, 0, stream>>>(
        Hh, W2t, N4, EMB, HID, nullptr, Gt, nullptr, nullptr, nullptr, nullptr, nullptr, nullptr);
    otgm_gemm_bf16<EPI_BIAS_BOTH><<<dim3(N4 / BM, EMB / BN), 256, 0, stream>>>(
        A1h, Gt, N4, EMB, N4, Z1f, Z1h, b2, nullptr, nullptr, nullptr, nullptr, nullptr);
    otgm_rownorm<<<N4, 256, 0, stream>>>(Z1f, nn1);

    // --- encoder, graph 2 ---
    otgm_gemm_bf16<EPI_BIAS_RELU><<<dim3(N4 / BM, HID / BN), 256, 0, stream>>>(
        A2h, W1t, N4, HID, N4, nullptr, Hh, b1, nullptr, nullptr, nullptr, nullptr, nullptr);
    otgm_gemm_bf16<EPI_HT><<<dim3(N4 / BM, EMB / BN), 256, 0, stream>>>(
        Hh, W2t, N4, EMB, HID, nullptr, Gt, nullptr, nullptr, nullptr, nullptr, nullptr, nullptr);
    otgm_gemm_bf16<EPI_BIAS_BOTH><<<dim3(N4 / BM, EMB / BN), 256, 0, stream>>>(
        A2h, Gt, N4, EMB, N4, Z2f, Z2h, b2, nullptr, nullptr, nullptr, nullptr, nullptr);
    otgm_rownorm<<<N4, 256, 0, stream>>>(Z2f, nn2);

    // --- fused sq-cdist + K = exp(-C/reg), writing K and K^T ---
    otgm_gemm_bf16<EPI_KEXP><<<dim3(N4 / BM, N4 / BN), 256, 0, stream>>>(
        Z1h, Z2h, N4, N4, EMB, Kmat, nullptr, nullptr, nn1, nn2, Ktm, nullptr, nullptr);

    // --- Sinkhorn (50 iters, L2-resident matvecs) ---
    otgm_fill<<<N4 / 256, 256, 0, stream>>>(vvec, N4, 1.f / N4);
    for (int it = 0; it < SINK_IT; ++it) {
        otgm_matvec<<<N4, 256, 0, stream>>>(Kmat, vvec, uvec, 1.f / N4);  // u = a/(K v)
        otgm_matvec<<<N4, 256, 0, stream>>>(Ktm, uvec, vvec, 1.f / N4);   // v = b/(K^T u)
    }

    // --- S = diag(u) K diag(v)  (f32 to d_out, bf16 for GEMMs) ---
    otgm_build_S<<<ELT_BLK, 256, 0, stream>>>(Kmat, uvec, vvec, outS, Sh);

    // --- loss = || M1 - (S@M2)@S^T ||_F ---
    otgm_gemm_bf16<EPI_H><<<dim3(N4 / BM, N4 / BN), 256, 0, stream>>>(
        Sh, M2t, N4, N4, N4, nullptr, Th, nullptr, nullptr, nullptr, nullptr, nullptr, nullptr);
    otgm_fill<<<1, 256, 0, stream>>>(lossAcc, 1, 0.f);
    otgm_gemm_bf16<EPI_LOSS><<<dim3(N4 / BM, N4 / BN), 256, 0, stream>>>(
        Th, Sh, N4, N4, N4, nullptr, nullptr, nullptr, nullptr, nullptr, nullptr, M1, lossAcc);
    otgm_finalize<<<1, 1, 0, stream>>>(lossAcc, outLoss);
}
